// CausualAttention_74758200754350
// MI455X (gfx1250) — compile-verified, matrix operand order checked by reading, not run on hardware
//
#include <hip/hip_runtime.h>
#include <hip/hip_bf16.h>

// ---------------------------------------------------------------------------
// CDNA5 / gfx1250 attention:  O = softmax((XW^T)(XW^T)^T / sqrt(d)) (XW^T)
// N = 8192, d_in = d_out = 1024, fp32 in/out, bf16 WMMA internally.
// ---------------------------------------------------------------------------

typedef __attribute__((ext_vector_type(16))) __bf16       v16bf;
typedef __attribute__((ext_vector_type(8)))  float        v8f;
typedef __attribute__((ext_vector_type(4)))  unsigned int u32x4;

#define NN 8192
#define DD 1024

union BFrag { u32x4 u[2]; v16bf v; };   // 32 bytes = 16 bf16 (one WMMA A/B frag)
union Acc   { v8f   v;    float f[8]; };

__device__ __forceinline__ unsigned short f2bf(float x) {
  unsigned u = __float_as_uint(x);
  u += 0x7fffu + ((u >> 16) & 1u);      // round-to-nearest-even
  return (unsigned short)(u >> 16);
}

// ------------------------- fp32 -> bf16 conversion -------------------------
__global__ __launch_bounds__(256) void cvt_kernel(const float* __restrict__ X,
                                                  const float* __restrict__ W,
                                                  unsigned short* __restrict__ Xb,
                                                  unsigned short* __restrict__ Wb) {
  const long long nx = (long long)NN * DD;
  const long long nw = (long long)DD * DD;
  long long i       = (long long)blockIdx.x * blockDim.x + threadIdx.x;
  long long stride  = (long long)gridDim.x * blockDim.x;
  for (; i < nx + nw; i += stride) {
    if (i < nx) Xb[i] = f2bf(X[i]);
    else        Wb[i - nx] = f2bf(W[i - nx]);
  }
}

// ------------------- projection: Q = X @ W^T  (bf16 WMMA) ------------------
// Writes Q row-major (Qb[8192][1024]) and transposed (Qt[1024][8192]).
__global__ __launch_bounds__(256) void proj_kernel(const unsigned short* __restrict__ Xb,
                                                   const unsigned short* __restrict__ Wb,
                                                   unsigned short* __restrict__ Qb,
                                                   unsigned short* __restrict__ Qt) {
  const int lane = threadIdx.x & 31;
  const int wave = threadIdx.x >> 5;
  const int l16  = lane & 15;
  const int kh   = (lane >> 4) << 3;            // 0 or 8 (K / M half select)

  const int wm = blockIdx.x * 128 + (wave & 3) * 32;   // 32x32 tile per wave
  const int wn = blockIdx.y * 64  + (wave >> 2) * 32;

  Acc acc[2][2];
#pragma unroll
  for (int mt = 0; mt < 2; ++mt)
#pragma unroll
    for (int nt = 0; nt < 2; ++nt)
#pragma unroll
      for (int r = 0; r < 8; ++r) acc[mt][nt].f[r] = 0.f;

  for (int k = 0; k < DD; k += 32) {
    BFrag a[2], b[2];
#pragma unroll
    for (int mt = 0; mt < 2; ++mt) {
      const unsigned short* p = Xb + (size_t)(wm + mt * 16 + l16) * DD + k + kh;
      a[mt].u[0] = *(const u32x4*)p;
      a[mt].u[1] = *(const u32x4*)(p + 16);
      __builtin_prefetch(p + 32, 0, 1);
    }
#pragma unroll
    for (int nt = 0; nt < 2; ++nt) {
      const unsigned short* p = Wb + (size_t)(wn + nt * 16 + l16) * DD + k + kh;
      b[nt].u[0] = *(const u32x4*)p;
      b[nt].u[1] = *(const u32x4*)(p + 16);
      __builtin_prefetch(p + 32, 0, 1);
    }
#pragma unroll
    for (int mt = 0; mt < 2; ++mt)
#pragma unroll
      for (int nt = 0; nt < 2; ++nt)
        acc[mt][nt].v = __builtin_amdgcn_wmma_f32_16x16x32_bf16(
            false, a[mt].v, false, b[nt].v, (short)0, acc[mt][nt].v, false, false);
  }

#pragma unroll
  for (int mt = 0; mt < 2; ++mt) {
    const int mbase = wm + mt * 16 + kh;        // C/D layout: lane half -> M half
#pragma unroll
    for (int nt = 0; nt < 2; ++nt) {
      const int n = wn + nt * 16 + l16;
#pragma unroll
      for (int r = 0; r < 8; ++r)               // row-major Q (16-bit scatter)
        Qb[(size_t)(mbase + r) * DD + n] = f2bf(acc[mt][nt].f[r]);
      u32x4 pk;                                 // transposed Q (one b128 store)
#pragma unroll
      for (int r = 0; r < 4; ++r)
        pk[r] = (unsigned int)f2bf(acc[mt][nt].f[2 * r]) |
                ((unsigned int)f2bf(acc[mt][nt].f[2 * r + 1]) << 16);
      *(u32x4*)(Qt + (size_t)n * NN + mbase) = pk;
    }
  }
}

// ----------------- flash attention (q = k = v = Q), fp32 out ---------------
// Block = 32 query rows x 8 waves; wave w owns D-slice [128w, 128w+128).
__global__ __launch_bounds__(256) void attn_kernel(const unsigned short* __restrict__ Qb,
                                                   const unsigned short* __restrict__ Qt,
                                                   float* __restrict__ Out) {
  const int lane  = threadIdx.x & 31;
  const int wave  = threadIdx.x >> 5;
  const int l16   = lane & 15;
  const int kh    = (lane >> 4) << 3;
  const int bm    = blockIdx.x * 32;            // query-row block
  const int dbase = wave * 128;                 // this wave's D slice

  __shared__ float          S_s[32 * 32];       // score tile (f32)
  __shared__ unsigned short P_s[32 * 32];       // exp tile (bf16)
  __shared__ float m_s[32], l_s[32], alpha_s[32];

  if (threadIdx.x < 32) {
    m_s[threadIdx.x] = -3.0e38f;
    l_s[threadIdx.x] = 0.f;
    alpha_s[threadIdx.x] = 1.f;
  }

  Acc oacc[2][8];
#pragma unroll
  for (int mt = 0; mt < 2; ++mt)
#pragma unroll
    for (int nt = 0; nt < 8; ++nt)
#pragma unroll
      for (int r = 0; r < 8; ++r) oacc[mt][nt].f[r] = 0.f;

  __syncthreads();

  for (int j = 0; j < NN; j += 32) {            // 256 key blocks of 32
    // zero the shared score tile (1024 f32 / 256 threads)
#pragma unroll
    for (int t = 0; t < 4; ++t) S_s[threadIdx.x * 4 + t] = 0.f;

    // partial scores over this wave's D slice: S += Q_blk * K_blk^T
    Acc sacc[2][2];
#pragma unroll
    for (int mt = 0; mt < 2; ++mt)
#pragma unroll
      for (int nt = 0; nt < 2; ++nt)
#pragma unroll
        for (int r = 0; r < 8; ++r) sacc[mt][nt].f[r] = 0.f;

#pragma unroll
    for (int kk = 0; kk < 128; kk += 32) {
      BFrag qa[2], kb[2];
#pragma unroll
      for (int mt = 0; mt < 2; ++mt) {
        const unsigned short* p = Qb + (size_t)(bm + mt * 16 + l16) * DD + dbase + kk + kh;
        qa[mt].u[0] = *(const u32x4*)p;
        qa[mt].u[1] = *(const u32x4*)(p + 16);
      }
#pragma unroll
      for (int nt = 0; nt < 2; ++nt) {
        const unsigned short* p = Qb + (size_t)(j + nt * 16 + l16) * DD + dbase + kk + kh;
        kb[nt].u[0] = *(const u32x4*)p;
        kb[nt].u[1] = *(const u32x4*)(p + 16);
      }
#pragma unroll
      for (int mt = 0; mt < 2; ++mt)
#pragma unroll
        for (int nt = 0; nt < 2; ++nt)
          sacc[mt][nt].v = __builtin_amdgcn_wmma_f32_16x16x32_bf16(
              false, qa[mt].v, false, kb[nt].v, (short)0, sacc[mt][nt].v, false, false);
    }

    __syncthreads();                            // zeroed tile visible
#pragma unroll
    for (int mt = 0; mt < 2; ++mt)
#pragma unroll
      for (int nt = 0; nt < 2; ++nt)
#pragma unroll
        for (int r = 0; r < 8; ++r)
          atomicAdd(&S_s[(mt * 16 + kh + r) * 32 + nt * 16 + l16], sacc[mt][nt].f[r]);
    __syncthreads();                            // full S available

    // wave 0: online softmax update, one lane per query row
    if (wave == 0) {
      const float scale = 0.03125f;             // 1/sqrt(1024)
      float mo = m_s[lane], lo = l_s[lane];
      float rmax = -3.0e38f;
#pragma unroll
      for (int c = 0; c < 32; ++c)
        rmax = fmaxf(rmax, S_s[lane * 32 + c] * scale);
      float mn    = fmaxf(mo, rmax);
      float alpha = __expf(mo - mn);
      float sum   = 0.f;
#pragma unroll
      for (int c = 0; c < 32; ++c) {
        float p = __expf(S_s[lane * 32 + c] * scale - mn);
        sum += p;
        P_s[lane * 32 + c] = f2bf(p);
      }
      m_s[lane]     = mn;
      l_s[lane]     = lo * alpha + sum;
      alpha_s[lane] = alpha;
    }
    __syncthreads();                            // P / alpha ready

    // rescale O and accumulate P @ V over this wave's D slice
    BFrag pa[2];
#pragma unroll
    for (int mt = 0; mt < 2; ++mt) {
      const unsigned short* p = &P_s[(mt * 16 + l16) * 32 + kh];
      pa[mt].u[0] = *(const u32x4*)p;           // ds_load_b128
      pa[mt].u[1] = *(const u32x4*)(p + 16);
    }
#pragma unroll
    for (int mt = 0; mt < 2; ++mt) {
      float al[8];
#pragma unroll
      for (int r = 0; r < 8; ++r) al[r] = alpha_s[mt * 16 + kh + r];
#pragma unroll
      for (int nt = 0; nt < 8; ++nt)
#pragma unroll
        for (int r = 0; r < 8; ++r) oacc[mt][nt].f[r] *= al[r];
    }
#pragma unroll
    for (int nt = 0; nt < 8; ++nt) {
      BFrag vb;                                 // V^T fragment from Qt (L2-resident)
      const unsigned short* p = Qt + (size_t)(dbase + nt * 16 + l16) * NN + j + kh;
      vb.u[0] = *(const u32x4*)p;
      vb.u[1] = *(const u32x4*)(p + 16);
      __builtin_prefetch(p + 32, 0, 1);
#pragma unroll
      for (int mt = 0; mt < 2; ++mt)
        oacc[mt][nt].v = __builtin_amdgcn_wmma_f32_16x16x32_bf16(
            false, pa[mt].v, false, vb.v, (short)0, oacc[mt][nt].v, false, false);
    }
  }

  // epilogue: divide by l and store fp32
#pragma unroll
  for (int mt = 0; mt < 2; ++mt) {
    float inv[8];
#pragma unroll
    for (int r = 0; r < 8; ++r) inv[r] = 1.0f / l_s[mt * 16 + kh + r];
#pragma unroll
    for (int nt = 0; nt < 8; ++nt) {
      const int col = dbase + nt * 16 + l16;
#pragma unroll
      for (int r = 0; r < 8; ++r)
        Out[(size_t)(bm + mt * 16 + kh + r) * DD + col] = oacc[mt][nt].f[r] * inv[r];
    }
  }
}

// ------------------------------- launcher ----------------------------------
extern "C" void kernel_launch(void* const* d_in, const int* in_sizes, int n_in,
                              void* d_out, int out_size, void* d_ws, size_t ws_size,
                              hipStream_t stream) {
  const float* X = (const float*)d_in[0];   // [8192, 1024]
  const float* W = (const float*)d_in[1];   // [1024, 1024]
  float* Out = (float*)d_out;               // [8192, 1024] fp32

  char* ws = (char*)d_ws;
  unsigned short* Xb = (unsigned short*)(ws);                           // 16 MB
  unsigned short* Wb = (unsigned short*)(ws + (size_t)16777216);        //  2 MB
  unsigned short* Qb = (unsigned short*)(ws + (size_t)18874368);        // 16 MB
  unsigned short* Qt = (unsigned short*)(ws + (size_t)35651584);        // 16 MB

  cvt_kernel<<<dim3(4608), dim3(256), 0, stream>>>(X, W, Xb, Wb);
  proj_kernel<<<dim3(64, 16), dim3(256), 0, stream>>>(Xb, Wb, Qb, Qt);
  attn_kernel<<<dim3(256), dim3(256), 0, stream>>>(Qb, Qt, Out);
}
